// encoder_layer_42631845380628
// MI455X (gfx1250) — compile-verified, matrix operand order checked by reading, not run on hardware
//
#include <hip/hip_runtime.h>
#include <math.h>

// Problem constants (match reference)
#define BB   32
#define SSZ  512
#define DD   256
#define HHD  4
#define DKK  64
#define DVV  64
#define DFF  1024

// Tiling: block = 256 threads = 8 wave32. Block tile 128(M) x 64(N), K-step 32.
// Waves arranged 4(M) x 2(N); each wave computes a 32x32 macro-tile = 4 WMMAs
// per K-step, reusing 2 A-fragments and 2 B-fragments.
// LDS rows padded to 56 u16 (112 B = 28-bank stride, 16B aligned) so the 16
// fragment rows map to distinct bank groups (conflict-free ds_load_b128).
#define LDSP 56

typedef __attribute__((ext_vector_type(16))) __bf16          v16bf;
typedef __attribute__((ext_vector_type(8)))  float           v8f;
typedef __attribute__((ext_vector_type(8)))  unsigned short  u16x8;
typedef __attribute__((ext_vector_type(4)))  unsigned short  u16x4;

union FragU { u16x8 u[2]; v16bf v; };

__device__ __forceinline__ unsigned short f2bf(float f) {
    unsigned int u = __float_as_uint(f);
    u += 0x7FFFu + ((u >> 16) & 1u);     // round-to-nearest-even
    return (unsigned short)(u >> 16);
}

__device__ __forceinline__ u16x4 f2bf4(float4 f) {
    u16x4 p;
    p[0] = f2bf(f.x); p[1] = f2bf(f.y); p[2] = f2bf(f.z); p[3] = f2bf(f.w);
    return p;
}

__device__ __forceinline__ float gelu_exact(float x) {
    return 0.5f * x * (1.0f + erff(x * 0.70710678118654752f));
}

// Shared inner compute: load 2 A-frags + 2 B-frags from LDS, do 4 WMMAs.
__device__ __forceinline__ void wmma_step(const unsigned short (*As)[LDSP],
                                          const unsigned short (*Bs)[LDSP],
                                          int wm, int wn, int lane, v8f acc[2][2])
{
    const int fr = lane & 15;
    const int kh = (lane >> 4) * 8;
    FragU fa[2], fb[2];
#pragma unroll
    for (int im = 0; im < 2; ++im) {
        const int row = wm * 32 + im * 16 + fr;
        fa[im].u[0] = *(const u16x8*)&As[row][kh];
        fa[im].u[1] = *(const u16x8*)&As[row][kh + 16];
    }
#pragma unroll
    for (int in = 0; in < 2; ++in) {
        const int col = wn * 32 + in * 16 + fr;
        fb[in].u[0] = *(const u16x8*)&Bs[col][kh];
        fb[in].u[1] = *(const u16x8*)&Bs[col][kh + 16];
    }
#pragma unroll
    for (int im = 0; im < 2; ++im)
#pragma unroll
        for (int in = 0; in < 2; ++in)
            acc[im][in] = __builtin_amdgcn_wmma_f32_16x16x32_bf16(
                false, fa[im].v, false, fb[in].v, (short)0, acc[im][in], false, false);
}

// ---------------------------------------------------------------------------
// Generic batched bf16-WMMA GEMM: C[z] = act(alpha * A[z] @ op(B[z]) + bias + addT[z])
// TA = float (fp32, converted on the fly) or unsigned short (pre-converted bf16).
// Requires: M % 128 == 0, N % 64 == 0, K % 32 == 0 (all launches satisfy this).
// Batch z decomposes as zo = z / binner, zi = z % binner with per-tensor strides.
// ---------------------------------------------------------------------------
template<typename TA, bool TRANSB, int ACT, bool ACCUM>
__global__ __launch_bounds__(256)
void gemm_wmma(const TA* __restrict__ A, long long aO, long long aI, int lda,
               const float* __restrict__ Bm, long long bO, long long bI, int ldb,
               const float* __restrict__ bias,
               const float* __restrict__ addT, long long adO, long long adI, int ldadd,
               float* __restrict__ C, long long cO, long long cI, int ldc,
               float alpha, int M, int N, int K, int binner)
{
    __shared__ __align__(16) unsigned short As[128][LDSP];
    __shared__ __align__(16) unsigned short Bs[64][LDSP];

    const int z  = blockIdx.z;
    const int zo = z / binner, zi = z % binner;
    A  += (long long)zo * aO + (long long)zi * aI;
    Bm += (long long)zo * bO + (long long)zi * bI;
    C  += (long long)zo * cO + (long long)zi * cI;
    if (addT) addT += (long long)zo * adO + (long long)zi * adI;

    const int tid  = threadIdx.x;
    const int lane = tid & 31;
    const int wid  = tid >> 5;
    const int wm   = wid >> 1;      // 0..3
    const int wn   = wid & 1;       // 0..1

    const int m0 = blockIdx.y * 128;
    const int n0 = blockIdx.x * 64;

    v8f acc[2][2] = {};

    for (int k0 = 0; k0 < K; k0 += 32) {
        __syncthreads();
        if constexpr (sizeof(TA) == 2) {
            // A already bf16: 128x32 u16 = 512 u16x8 chunks, 2 per thread
#pragma unroll
            for (int t = 0; t < 2; ++t) {
                int qi = tid + t * 256;
                int r = qi >> 2, c = (qi & 3) << 3;
                *(u16x8*)&As[r][c] =
                    *(const u16x8*)((const unsigned short*)A + (long long)(m0 + r) * lda + (k0 + c));
            }
        } else {
            // A fp32: 128x32 = 1024 float4, 4 per thread, convert to bf16
#pragma unroll
            for (int t = 0; t < 4; ++t) {
                int qi = tid + t * 256;
                int r = qi >> 3, c = (qi & 7) << 2;
                float4 f = *(const float4*)((const float*)A + (long long)(m0 + r) * lda + (k0 + c));
                *(u16x4*)&As[r][c] = f2bf4(f);
            }
        }
        // B tile -> Bs[n][k]: 64x32 = 512 float4, 2 per thread
#pragma unroll
        for (int t = 0; t < 2; ++t) {
            int qi = tid + t * 256;
            if (TRANSB) {                        // B given as [N, K]
                int n = qi >> 3, c = (qi & 7) << 2;
                float4 f = *(const float4*)(Bm + (long long)(n0 + n) * ldb + (k0 + c));
                *(u16x4*)&Bs[n][c] = f2bf4(f);
            } else {                             // B given as [K, N]
                int kk = qi >> 4, n = (qi & 15) << 2;
                float4 f = *(const float4*)(Bm + (long long)(k0 + kk) * ldb + (n0 + n));
                Bs[n + 0][kk] = f2bf(f.x);
                Bs[n + 1][kk] = f2bf(f.y);
                Bs[n + 2][kk] = f2bf(f.z);
                Bs[n + 3][kk] = f2bf(f.w);
            }
        }
        __syncthreads();
        wmma_step(As, Bs, wm, wn, lane, acc);
    }

    const int mrr = (lane >> 4) << 3;
#pragma unroll
    for (int in = 0; in < 2; ++in) {
        const int nc = n0 + wn * 32 + in * 16 + (lane & 15);
        const float bval = bias ? bias[nc] : 0.f;
#pragma unroll
        for (int im = 0; im < 2; ++im) {
#pragma unroll
            for (int r = 0; r < 8; ++r) {
                int mr = m0 + wm * 32 + im * 16 + r + mrr;
                float vv = acc[im][in][r] * alpha + bval;
                if (addT) vv += addT[(long long)mr * ldadd + nc];
                if (ACT == 1) vv = gelu_exact(vv);
                long long idx = (long long)mr * ldc + nc;
                if (ACCUM) C[idx] += vv; else C[idx] = vv;
            }
        }
    }
}

// ---------------------------------------------------------------------------
// Relation scores: for fixed i, scores[b,h,i,j] += scale * sum_d q[b,i,h,d]*ek[i,j,d]
// A = Q_all[i] : 128 rows r=(b*H+h), K=64 ; B = edges_key[i] : [S, DK] (NT form)
// grid: (S/64 j-tiles, 1, S i-values); block tile 128 x 64.
// ---------------------------------------------------------------------------
__global__ __launch_bounds__(256)
void rel_scores_wmma(const float* __restrict__ q, const float* __restrict__ ek,
                     float* __restrict__ scores, float scale)
{
    __shared__ __align__(16) unsigned short As[128][LDSP];
    __shared__ __align__(16) unsigned short Bs[64][LDSP];

    const int i    = blockIdx.z;
    const int tid  = threadIdx.x;
    const int lane = tid & 31;
    const int wid  = tid >> 5;
    const int wm   = wid >> 1, wn = wid & 1;
    const int j0   = blockIdx.x * 64;

    v8f acc[2][2] = {};

#pragma unroll
    for (int k0 = 0; k0 < DKK; k0 += 32) {
        __syncthreads();
#pragma unroll
        for (int t = 0; t < 4; ++t) {
            int qi = tid + t * 256;
            int r = qi >> 3, c = (qi & 7) << 2;
            int b = r >> 2, h = r & 3;                   // H == 4
            float4 f = *(const float4*)(q + (long long)b * (SSZ * HHD * DKK) +
                                        (long long)i * (HHD * DKK) + h * DKK + k0 + c);
            *(u16x4*)&As[r][c] = f2bf4(f);
        }
#pragma unroll
        for (int t = 0; t < 2; ++t) {
            int qi = tid + t * 256;
            int n = qi >> 3, c = (qi & 7) << 2;
            float4 f = *(const float4*)(ek + ((long long)i * SSZ + (j0 + n)) * DKK + k0 + c);
            *(u16x4*)&Bs[n][c] = f2bf4(f);
        }
        __syncthreads();
        wmma_step(As, Bs, wm, wn, lane, acc);
    }

    const int mrr = (lane >> 4) << 3;
#pragma unroll
    for (int in = 0; in < 2; ++in) {
        const int j = j0 + wn * 32 + in * 16 + (lane & 15);
#pragma unroll
        for (int im = 0; im < 2; ++im) {
#pragma unroll
            for (int r = 0; r < 8; ++r) {
                int rr = wm * 32 + im * 16 + r + mrr;
                scores[((long long)rr * SSZ + i) * SSZ + j] += acc[im][in][r] * scale;
            }
        }
    }
}

// ---------------------------------------------------------------------------
// Relation context: for fixed i, ctx[b,i,h,d] += sum_j W[b,h,i,j] * ev[i,j,d]
// A rows r=(b*H+h): A[r][j] = Wbf[r*S*S + i*S + j] (bf16 softmax weights)
// B = edges_value[i] : [S, DV]. grid: (1, 1, S); block tile 128 x 64, K = 512.
// ---------------------------------------------------------------------------
__global__ __launch_bounds__(256)
void rel_ctx_wmma(const unsigned short* __restrict__ w, const float* __restrict__ ev,
                  float* __restrict__ ctx)
{
    __shared__ __align__(16) unsigned short As[128][LDSP];
    __shared__ __align__(16) unsigned short Bs[64][LDSP];

    const int i    = blockIdx.z;
    const int tid  = threadIdx.x;
    const int lane = tid & 31;
    const int wid  = tid >> 5;
    const int wm   = wid >> 1, wn = wid & 1;

    v8f acc[2][2] = {};

    for (int k0 = 0; k0 < SSZ; k0 += 32) {
        __syncthreads();
        // A tile (bf16 pass-through): 128x32 u16 = 512 u16x8 chunks, 2 per thread
#pragma unroll
        for (int t = 0; t < 2; ++t) {
            int qi = tid + t * 256;
            int r = qi >> 2, c = (qi & 3) << 3;
            *(u16x8*)&As[r][c] =
                *(const u16x8*)(w + (long long)r * (SSZ * SSZ) + (long long)i * SSZ + k0 + c);
        }
#pragma unroll
        for (int t = 0; t < 2; ++t) {
            int qi = tid + t * 256;
            int kk = qi >> 4, n = (qi & 15) << 2;
            float4 f = *(const float4*)(ev + ((long long)i * SSZ + k0 + kk) * DVV + n);
            Bs[n + 0][kk] = f2bf(f.x);
            Bs[n + 1][kk] = f2bf(f.y);
            Bs[n + 2][kk] = f2bf(f.z);
            Bs[n + 3][kk] = f2bf(f.w);
        }
        __syncthreads();
        wmma_step(As, Bs, wm, wn, lane, acc);
    }

    const int mrr = (lane >> 4) << 3;
#pragma unroll
    for (int in = 0; in < 2; ++in) {
        const int d = wn * 32 + in * 16 + (lane & 15);
#pragma unroll
        for (int im = 0; im < 2; ++im) {
#pragma unroll
            for (int r = 0; r < 8; ++r) {
                int rr = wm * 32 + im * 16 + r + mrr;
                int b = rr >> 2, h = rr & 3;
                ctx[(long long)b * (SSZ * HHD * DVV) + (long long)i * (HHD * DVV) +
                    h * DVV + d] += acc[im][in][r];
            }
        }
    }
}

// ---------------------------------------------------------------------------
// Row softmax over last dim (fixed 512), fp32 in -> bf16 out.
// One wave32 per row; each lane owns 16 contiguous columns (4x float4 loads,
// 2x u16x8 stores).
// ---------------------------------------------------------------------------
__global__ __launch_bounds__(256)
void softmax_rows(const float* __restrict__ s, unsigned short* __restrict__ wout,
                  int rows)
{
    const int wid  = threadIdx.x >> 5;
    const int lane = threadIdx.x & 31;
    const int row  = blockIdx.x * 8 + wid;
    if (row >= rows) return;
    const float* p = s + (long long)row * SSZ + lane * 16;

    float vals[16];
#pragma unroll
    for (int t = 0; t < 4; ++t) {
        float4 f = *(const float4*)(p + t * 4);
        vals[t * 4 + 0] = f.x; vals[t * 4 + 1] = f.y;
        vals[t * 4 + 2] = f.z; vals[t * 4 + 3] = f.w;
    }
    float m = -3.0e38f;
#pragma unroll
    for (int l = 0; l < 16; ++l) m = fmaxf(m, vals[l]);
#pragma unroll
    for (int d = 16; d >= 1; d >>= 1) m = fmaxf(m, __shfl_xor(m, d, 32));
    float sum = 0.f;
#pragma unroll
    for (int l = 0; l < 16; ++l) { vals[l] = __expf(vals[l] - m); sum += vals[l]; }
#pragma unroll
    for (int d = 16; d >= 1; d >>= 1) sum += __shfl_xor(sum, d, 32);
    const float inv = 1.f / sum;

    unsigned short* po = wout + (long long)row * SSZ + lane * 16;
#pragma unroll
    for (int t = 0; t < 2; ++t) {
        u16x8 pk;
#pragma unroll
        for (int l = 0; l < 8; ++l) pk[l] = f2bf(vals[t * 8 + l] * inv);
        *(u16x8*)(po + t * 8) = pk;
    }
}

// ---------------------------------------------------------------------------
// out = LayerNorm(a + res) * g + b  over D=256. One wave32 per row, 8 / lane.
// ---------------------------------------------------------------------------
__global__ __launch_bounds__(256)
void add_ln(const float* __restrict__ a, const float* __restrict__ res,
            const float* __restrict__ g, const float* __restrict__ bt,
            float* __restrict__ out, int rows)
{
    const int wid  = threadIdx.x >> 5;
    const int lane = threadIdx.x & 31;
    const int row  = blockIdx.x * 8 + wid;
    if (row >= rows) return;
    const float* pa = a + (long long)row * DD;
    const float* pr = res + (long long)row * DD;

    float v[8];
    float s = 0.f;
#pragma unroll
    for (int l = 0; l < 8; ++l) { int c = lane + l * 32; v[l] = pa[c] + pr[c]; s += v[l]; }
#pragma unroll
    for (int d = 16; d >= 1; d >>= 1) s += __shfl_xor(s, d, 32);
    const float mean = s * (1.f / 256.f);
    float q = 0.f;
#pragma unroll
    for (int l = 0; l < 8; ++l) { float dlt = v[l] - mean; q += dlt * dlt; }
#pragma unroll
    for (int d = 16; d >= 1; d >>= 1) q += __shfl_xor(q, d, 32);
    const float inv = rsqrtf(q * (1.f / 256.f) + 1e-7f);
#pragma unroll
    for (int l = 0; l < 8; ++l) {
        int c = lane + l * 32;
        out[(long long)row * DD + c] = (v[l] - mean) * inv * g[c] + bt[c];
    }
}

// ---------------------------------------------------------------------------
extern "C" void kernel_launch(void* const* d_in, const int* in_sizes, int n_in,
                              void* d_out, int out_size, void* d_ws, size_t ws_size,
                              hipStream_t stream)
{
    (void)in_sizes; (void)n_in; (void)out_size; (void)ws_size;

    const float* enc  = (const float*)d_in[0];
    const float* ek   = (const float*)d_in[1];
    const float* ev   = (const float*)d_in[2];
    const float* abia = (const float*)d_in[3];
    const float* Wq   = (const float*)d_in[4];
    const float* bq   = (const float*)d_in[5];
    const float* Wk   = (const float*)d_in[6];
    const float* bk   = (const float*)d_in[7];
    const float* Wv   = (const float*)d_in[8];
    const float* bv   = (const float*)d_in[9];
    const float* Wo   = (const float*)d_in[10];
    const float* bo   = (const float*)d_in[11];
    const float* g1   = (const float*)d_in[12];
    const float* b1   = (const float*)d_in[13];
    const float* W1   = (const float*)d_in[14];
    const float* bf1  = (const float*)d_in[15];
    const float* W2   = (const float*)d_in[16];
    const float* bf2  = (const float*)d_in[17];
    const float* g2   = (const float*)d_in[18];
    const float* b2   = (const float*)d_in[19];
    float* out = (float*)d_out;

    const size_t NTOK = (size_t)BB * SSZ;               // 16384
    const size_t BHSS = (size_t)BB * HHD * SSZ * SSZ;   // 33.5M
    float* ws     = (float*)d_ws;
    float* q      = ws;                                 // [B,S,H,DK]
    float* k      = q    + NTOK * 256;                  // [B,S,H,DK]
    float* v      = k    + NTOK * 256;                  // [B,S,H,DV]
    float* scores = v    + NTOK * 256;                  // [B,H,S,S] fp32 logits
    unsigned short* wbf = (unsigned short*)(scores + BHSS);  // [B,H,S,S] bf16 weights
    float* ctx    = (float*)(wbf + BHSS);               // [B,S,H,DV]
    float* attn   = ctx  + NTOK * 256;                  // [B,S,D]
    float* xbuf   = attn + NTOK * 256;                  // [B,S,D]
    float* hbuf   = xbuf + NTOK * 256;                  // [B,S,DFF]
    float* ff2    = hbuf + NTOK * DFF;                  // [B,S,D]

    const dim3 blk(256);
    const float scale = 0.125f;                         // 1/sqrt(64)

    // 1-3) Q, K, V projections: [16384,256] @ [256,256] + bias
    {
        dim3 grd(DD / 64, NTOK / 128, 1);
        gemm_wmma<float, false, 0, false><<<grd, blk, 0, stream>>>(
            enc, 0, 0, DD,  Wq, 0, 0, DD,  bq,
            nullptr, 0, 0, 0,  q, 0, 0, HHD * DKK,
            1.0f, (int)NTOK, HHD * DKK, DD, 1);
        gemm_wmma<float, false, 0, false><<<grd, blk, 0, stream>>>(
            enc, 0, 0, DD,  Wk, 0, 0, DD,  bk,
            nullptr, 0, 0, 0,  k, 0, 0, HHD * DKK,
            1.0f, (int)NTOK, HHD * DKK, DD, 1);
        gemm_wmma<float, false, 0, false><<<grd, blk, 0, stream>>>(
            enc, 0, 0, DD,  Wv, 0, 0, DD,  bv,
            nullptr, 0, 0, 0,  v, 0, 0, HHD * DVV,
            1.0f, (int)NTOK, HHD * DVV, DD, 1);
    }

    // 4) Content scores: scores[b,h] = scale * q[b,h] @ k[b,h]^T + attn_bias[b,h]
    {
        dim3 grd(SSZ / 64, SSZ / 128, BB * HHD);
        gemm_wmma<float, true, 0, false><<<grd, blk, 0, stream>>>(
            q, (long long)SSZ * HHD * DKK, DKK, HHD * DKK,
            k, (long long)SSZ * HHD * DKK, DKK, HHD * DKK,
            nullptr,
            abia, (long long)HHD * SSZ * SSZ, (long long)SSZ * SSZ, SSZ,
            scores, (long long)HHD * SSZ * SSZ, (long long)SSZ * SSZ, SSZ,
            scale, SSZ, SSZ, DKK, HHD);
    }

    // 5) Relation scores: scores[b,h,i,:] += scale * q[b,i,h,:] @ edges_key[i]^T
    {
        dim3 grd(SSZ / 64, 1, SSZ);
        rel_scores_wmma<<<grd, blk, 0, stream>>>(q, ek, scores, scale);
    }

    // 6) Softmax over j: fp32 logits -> bf16 weights
    {
        int rows = BB * HHD * SSZ;
        softmax_rows<<<dim3(rows / 8), blk, 0, stream>>>(scores, wbf, rows);
    }

    // 7) Content context: ctx[b,:,h,:] = Wbf[b,h] @ v[b,:,h,:]   (bf16 A path)
    {
        dim3 grd(DVV / 64, SSZ / 128, BB * HHD);
        gemm_wmma<unsigned short, false, 0, false><<<grd, blk, 0, stream>>>(
            wbf, (long long)HHD * SSZ * SSZ, (long long)SSZ * SSZ, SSZ,
            v, (long long)SSZ * HHD * DVV, DVV, HHD * DVV,
            nullptr,  nullptr, 0, 0, 0,
            ctx, (long long)SSZ * HHD * DVV, DVV, HHD * DVV,
            1.0f, SSZ, DVV, SSZ, HHD);
    }

    // 8) Relation context: ctx[b,i,h,:] += Wbf[b,h,i,:] @ edges_value[i]
    {
        dim3 grd(1, 1, SSZ);
        rel_ctx_wmma<<<grd, blk, 0, stream>>>(wbf, ev, ctx);
    }

    // 9) Output projection: attn = ctx_flat @ Wo + bo
    {
        dim3 grd(DD / 64, NTOK / 128, 1);
        gemm_wmma<float, false, 0, false><<<grd, blk, 0, stream>>>(
            ctx, 0, 0, HHD * DVV,  Wo, 0, 0, DD,  bo,
            nullptr, 0, 0, 0,  attn, 0, 0, DD,
            1.0f, (int)NTOK, DD, HHD * DVV, 1);
    }

    // 10) x = LN(enc + attn)
    add_ln<<<dim3(NTOK / 8), blk, 0, stream>>>(enc, attn, g1, b1, xbuf, (int)NTOK);

    // 11) h = GELU(x @ W1 + bf1)
    {
        dim3 grd(DFF / 64, NTOK / 128, 1);
        gemm_wmma<float, false, 1, false><<<grd, blk, 0, stream>>>(
            xbuf, 0, 0, DD,  W1, 0, 0, DFF,  bf1,
            nullptr, 0, 0, 0,  hbuf, 0, 0, DFF,
            1.0f, (int)NTOK, DFF, DD, 1);
    }

    // 12) ff2 = h @ W2 + bf2
    {
        dim3 grd(DD / 64, NTOK / 128, 1);
        gemm_wmma<float, false, 0, false><<<grd, blk, 0, stream>>>(
            hbuf, 0, 0, DFF,  W2, 0, 0, DD,  bf2,
            nullptr, 0, 0, 0,  ff2, 0, 0, DD,
            1.0f, (int)NTOK, DD, DFF, 1);
    }

    // 13) out = LN(x + ff2)
    add_ln<<<dim3(NTOK / 8), blk, 0, stream>>>(xbuf, ff2, g2, b2, out, (int)NTOK);
}
